// DomainSpecificHeads_28939489640780
// MI455X (gfx1250) — matrix operand order repaired and checked by reading, not compiled, not run on hardware
//
#include <hip/hip_runtime.h>

typedef __attribute__((ext_vector_type(16))) __bf16 v16bf;
typedef __attribute__((ext_vector_type(8)))  __bf16 v8bf;
typedef __attribute__((ext_vector_type(4)))  __bf16 v4bf;
typedef __attribute__((ext_vector_type(2)))  __bf16 v2bf;
typedef __attribute__((ext_vector_type(8)))  float  v8f;
typedef __attribute__((ext_vector_type(4)))  float  v4f;

#define TILE_M 128
#define TILE_N 128
#define TILE_K 32
#define LDSK   40   // padded K-stride (80B): 16B-aligned, conflict-free half-wave access

// Assemble a 16x32 bf16 A-frag / 32x16 bf16 B-frag lane vector from two
// 16-byte LDS reads (ds_load_b128 x2).
__device__ __forceinline__ v16bf make_frag(const __bf16* p0, const __bf16* p1) {
    v8bf lo = *(const v8bf*)p0;
    v8bf hi = *(const v8bf*)p1;
    return __builtin_shufflevector(lo, hi, 0,1,2,3,4,5,6,7,8,9,10,11,12,13,14,15);
}

// ---------------------------------------------------------------------------
// GEMM1: hidden = hs[4096,1024] @ W_base[1024,1024] + b_base  -> bf16 in ws
// ---------------------------------------------------------------------------
__global__ __launch_bounds__(256)
void dsh_base_proj_kernel(const float* __restrict__ A,     // [4096,1024] f32
                          const float* __restrict__ Bm,    // [1024,1024] f32
                          const float* __restrict__ bias,  // [1024] f32
                          __bf16*      __restrict__ Hout)  // [4096,1024] bf16
{
    __shared__ __align__(16) __bf16 lds_a[TILE_M][LDSK];  // [m][k]
    __shared__ __align__(16) __bf16 lds_b[TILE_N][LDSK];  // transposed: [n][k]

    const int tid   = threadIdx.x;
    const int lane  = tid & 31;
    const int wave  = tid >> 5;
    const int waveM = wave >> 1;      // 0..3  -> 32-row strip
    const int waveN = wave & 1;       // 0..1  -> 64-col strip
    const int m0 = blockIdx.y * TILE_M;
    const int n0 = blockIdx.x * TILE_N;
    const int K = 1024, lda = 1024, ldb = 1024, ldc = 1024;

    const int l15  = lane & 15;
    const int half = lane >> 4;       // K-half select per ISA 16-bit layouts
    const int a_kb = half * 8;
    const int b_kb = half * 16;

    v8f acc[2][4] = {};

    for (int k0 = 0; k0 < K; k0 += TILE_K) {
        // Stage A tile 128x32 f32 -> bf16; packed 8B LDS stores
        #pragma unroll
        for (int i = 0; i < 4; ++i) {
            int linear = tid + i * 256;          // vec4 id, 1024 total
            int row = linear >> 3;               // 8 vec4 per 32-elem row
            int c4  = (linear & 7) << 2;
            v4f v = *(const v4f*)&A[(size_t)(m0 + row) * lda + k0 + c4];
            v4bf p = { (__bf16)v[0], (__bf16)v[1], (__bf16)v[2], (__bf16)v[3] };
            *(v4bf*)&lds_a[row][c4] = p;         // ds_store_b64
        }
        // Stage B tile 32x128 f32 -> bf16 transposed [n][k]; each thread does a
        // 2(K)x4(N) patch so column stores are packed K-pairs (ds_store_b32).
        #pragma unroll
        for (int i = 0; i < 2; ++i) {
            int linear = tid + i * 256;          // patch id, 512 total
            int k  = (linear >> 5) << 1;         // even K row
            int c4 = (linear & 31) << 2;
            v4f v0 = *(const v4f*)&Bm[(size_t)(k0 + k)     * ldb + n0 + c4];
            v4f v1 = *(const v4f*)&Bm[(size_t)(k0 + k + 1) * ldb + n0 + c4];
            #pragma unroll
            for (int j = 0; j < 4; ++j) {
                v2bf p = { (__bf16)v0[j], (__bf16)v1[j] };
                *(v2bf*)&lds_b[c4 + j][k] = p;   // ds_store_b32
            }
        }
        __syncthreads();

        v16bf afrag[2], bfrag[4];
        #pragma unroll
        for (int mi = 0; mi < 2; ++mi) {
            const int r = waveM * 32 + mi * 16 + l15;
            afrag[mi] = make_frag(&lds_a[r][a_kb], &lds_a[r][16 + a_kb]);
        }
        #pragma unroll
        for (int ni = 0; ni < 4; ++ni) {
            const int c = waveN * 64 + ni * 16 + l15;
            bfrag[ni] = make_frag(&lds_b[c][a_kb], &lds_b[c][16 + a_kb]);
        }
        #pragma unroll
        for (int mi = 0; mi < 2; ++mi)
            #pragma unroll
            for (int ni = 0; ni < 4; ++ni)
                acc[mi][ni] = __builtin_amdgcn_wmma_f32_16x16x32_bf16(
                    false, afrag[mi], false, bfrag[ni],
                    (short)0, acc[mi][ni], false, false);
        __syncthreads();
    }

    // Epilogue: + bias, convert to bf16, store
    #pragma unroll
    for (int mi = 0; mi < 2; ++mi) {
        #pragma unroll
        for (int ni = 0; ni < 4; ++ni) {
            const int col  = n0 + waveN * 64 + ni * 16 + l15;
            const int rowb = m0 + waveM * 32 + mi * 16 + half * 8;
            const float bv = bias[col];
            #pragma unroll
            for (int e = 0; e < 8; ++e)
                Hout[(size_t)(rowb + e) * ldc + col] = (__bf16)(acc[mi][ni][e] + bv);
        }
    }
}

// ---------------------------------------------------------------------------
// GEMM2: out[b] = hidden_bf16[b][512,1024] @ W_heads[idx_b][1024,32000] + bias
// A tile staged with GLOBAL_LOAD_ASYNC_TO_LDS_B128 (ASYNCcnt, no VGPR bounce).
// ---------------------------------------------------------------------------
__global__ __launch_bounds__(256)
void dsh_head_gemm_kernel(const __bf16* __restrict__ H,        // [8,512,1024] bf16
                          const int*    __restrict__ domain_ids,
                          const float*  __restrict__ W_heads,  // [9,1024,32000] f32
                          const float*  __restrict__ b_heads,  // [9,32000] f32
                          float*        __restrict__ Out)      // [8,512,32000] f32
{
    __shared__ __align__(16) __bf16 lds_a[TILE_M][LDSK];
    __shared__ __align__(16) __bf16 lds_b[TILE_N][LDSK];

    const int batch = blockIdx.z;
    const int did   = domain_ids[batch];
    const int idx   = (did >= 0 && did < 8) ? did : 8;   // default head = slot ND

    const __bf16* A    = H       + (size_t)batch * 512 * 1024;
    const float*  W    = W_heads + (size_t)idx * 1024 * 32000;
    const float*  bias = b_heads + (size_t)idx * 32000;
    float*        C    = Out     + (size_t)batch * 512 * 32000;

    const int tid   = threadIdx.x;
    const int lane  = tid & 31;
    const int wave  = tid >> 5;
    const int waveM = wave >> 1;
    const int waveN = wave & 1;
    const int m0 = blockIdx.y * TILE_M;
    const int n0 = blockIdx.x * TILE_N;
    const int K = 1024, lda = 1024, ldb = 32000, ldc = 32000;

    const int l15  = lane & 15;
    const int half = lane >> 4;
    const int a_kb = half * 8;
    const int b_kb = half * 16;

    v8f acc[2][4] = {};

    for (int k0 = 0; k0 < K; k0 += TILE_K) {
        // Stage A tile 128x32 bf16: async DMA global->LDS, 16B per lane.
        // GVS mode: 64-bit SGPR base + 32-bit per-lane byte offset; VDST VGPR
        // carries the LDS byte offset (flat LDS address truncates to [31:0]).
        #pragma unroll
        for (int i = 0; i < 2; ++i) {
            int linear = tid + i * 256;          // 8-elem group id, 512 total
            int row = linear >> 2;               // 4 groups per 32-elem row
            int g   = (linear & 3) << 3;
            unsigned gofs = (unsigned)(((m0 + row) * lda + k0 + g) * 2);
            unsigned lofs = (unsigned)(unsigned long long)&lds_a[row][g];
            asm volatile("global_load_async_to_lds_b128 %0, %1, %2"
                         :: "v"(lofs), "v"(gofs), "s"(A)
                         : "memory");
        }
        // Stage W tile 32x128 f32 -> bf16 transposed, packed K-pair stores;
        // prefetch next K-tile of the dominant HBM stream.
        #pragma unroll
        for (int i = 0; i < 2; ++i) {
            int linear = tid + i * 256;          // patch id, 512 total
            int k  = (linear >> 5) << 1;
            int c4 = (linear & 31) << 2;
            const float* src0 = &W[(size_t)(k0 + k) * ldb + n0 + c4];
            v4f v0 = *(const v4f*)src0;
            v4f v1 = *(const v4f*)(src0 + ldb);
            if (k0 + TILE_K < K)
                __builtin_prefetch(src0 + (size_t)TILE_K * ldb, 0, 1);
            #pragma unroll
            for (int j = 0; j < 4; ++j) {
                v2bf p = { (__bf16)v0[j], (__bf16)v1[j] };
                *(v2bf*)&lds_b[c4 + j][k] = p;
            }
        }
        asm volatile("s_wait_asynccnt 0x0" ::: "memory");
        __syncthreads();

        v16bf afrag[2], bfrag[4];
        #pragma unroll
        for (int mi = 0; mi < 2; ++mi) {
            const int r = waveM * 32 + mi * 16 + l15;
            afrag[mi] = make_frag(&lds_a[r][a_kb], &lds_a[r][16 + a_kb]);
        }
        #pragma unroll
        for (int ni = 0; ni < 4; ++ni) {
            const int c = waveN * 64 + ni * 16 + l15;
            bfrag[ni] = make_frag(&lds_b[c][a_kb], &lds_b[c][16 + a_kb]);
        }
        #pragma unroll
        for (int mi = 0; mi < 2; ++mi)
            #pragma unroll
            for (int ni = 0; ni < 4; ++ni)
                acc[mi][ni] = __builtin_amdgcn_wmma_f32_16x16x32_bf16(
                    false, afrag[mi], false, bfrag[ni],
                    (short)0, acc[mi][ni], false, false);
        __syncthreads();
    }

    #pragma unroll
    for (int mi = 0; mi < 2; ++mi) {
        #pragma unroll
        for (int ni = 0; ni < 4; ++ni) {
            const int col  = n0 + waveN * 64 + ni * 16 + l15;
            const int rowb = m0 + waveM * 32 + mi * 16 + half * 8;
            const float bv = bias[col];
            #pragma unroll
            for (int e = 0; e < 8; ++e)
                C[(size_t)(rowb + e) * ldc + col] = acc[mi][ni][e] + bv;
        }
    }
}

extern "C" void kernel_launch(void* const* d_in, const int* in_sizes, int n_in,
                              void* d_out, int out_size, void* d_ws, size_t ws_size,
                              hipStream_t stream) {
    const float* hs  = (const float*)d_in[0];   // [8,512,1024]
    const int*   dom = (const int*)  d_in[1];   // [8]
    const float* Wb  = (const float*)d_in[2];   // [1024,1024]
    const float* bb  = (const float*)d_in[3];   // [1024]
    const float* Wh  = (const float*)d_in[4];   // [9,1024,32000]
    const float* bh  = (const float*)d_in[5];   // [9,32000]
    float*       out = (float*)d_out;           // [8,512,32000]
    __bf16*      hbf = (__bf16*)d_ws;           // 4096*1024 bf16 = 8 MB scratch

    dim3 g1(1024 / TILE_N, 4096 / TILE_M);          // (8, 32)
    dsh_base_proj_kernel<<<g1, 256, 0, stream>>>(hs, Wb, bb, hbf);

    dim3 g2(32000 / TILE_N, 512 / TILE_M, 8);       // (250, 4, 8)
    dsh_head_gemm_kernel<<<g2, 256, 0, stream>>>(hbf, dom, Wh, bh, out);
}
